// RWKV_BiDir_x080_40157944217755
// MI455X (gfx1250) — matrix operand order repaired and checked by reading, not compiled, not run on hardware
//
#include <hip/hip_runtime.h>
#include <hip/hip_bf16.h>

// ---------------- problem constants (from the reference) ----------------
constexpr int kB = 2, kT = 2048, kC = 1024, kL = 6, kH = 16, kN = 64;
constexpr int kV = 16384, kFF = 4096;
constexpr int kM = kB * kT; // 4096 token rows

// ---------------- WMMA fragment types (CDNA5 wave32) ----------------
typedef __attribute__((ext_vector_type(16))) __bf16 v16bf;
typedef __attribute__((ext_vector_type(8)))  float  v8f;

union FragBf {
    v16bf v;
    uint4 q[2];
};

// ---------------- TDM descriptor helpers (CDNA5 Tensor Data Mover) ------
typedef __attribute__((ext_vector_type(4))) unsigned int u32x4_t;
typedef __attribute__((ext_vector_type(8))) int          i32x8_t;
typedef __attribute__((ext_vector_type(4))) int          i32x4_t;

union TdmG0 { unsigned long long w[2]; u32x4_t v; };
union TdmG1 { unsigned long long w[4]; i32x8_t v; };

// 2D tile load: tile_dim0 elems per row (contiguous), tile_rows rows,
// row stride in elements, element size 2 bytes (bf16), optional LDS row pad.
__device__ __forceinline__ void tdm_load_2d_bf16(unsigned lds_off,
                                                 const void* gaddr,
                                                 unsigned tile_x, unsigned tile_rows,
                                                 unsigned long long stride_elems,
                                                 int pad_interval_code,
                                                 int pad_amount_code) {
    TdmG0 g0;
    g0.w[0] = 1ull /*count=1*/ | ((unsigned long long)lds_off << 32);
    g0.w[1] = ((unsigned long long)gaddr & 0x01FFFFFFFFFFFFFFull) | (2ull << 62) /*type=2*/;

    unsigned d0 = (1u << 16); // data_size = 1 -> 2 bytes
    if (pad_amount_code >= 0)
        d0 |= (1u << 20) | ((unsigned)pad_interval_code << 22)
            | ((unsigned)pad_amount_code << 25);
    const unsigned long long td0 = 0x40000000ull; // tensor_dim0 (huge, no OOB)
    const unsigned long long td1 = 0x40000000ull; // tensor_dim1
    TdmG1 g1;
    g1.w[0] = (unsigned long long)d0 | (td0 << 48);
    g1.w[1] = (td0 >> 16) | (td1 << 16) | ((unsigned long long)tile_x << 48);
    g1.w[2] = (unsigned long long)tile_rows /*tile_dim1*/
            | ((stride_elems & 0xFFFFFFFFull) << 32);
    g1.w[3] = (stride_elems >> 32) & 0xFFFFull; // tensor_dim1_stride = 0
    i32x4_t zz = {0, 0, 0, 0};
    i32x8_t z8 = {0, 0, 0, 0, 0, 0, 0, 0};
    __builtin_amdgcn_tensor_load_to_lds(g0.v, g1.v, zz, zz, z8, 0);
}

// ======================================================================
// elementwise / helper kernels
// ======================================================================
__global__ void k_zero_f32(float* __restrict__ p, int n) {
    int i = blockIdx.x * 256 + threadIdx.x;
    if (i < n) p[i] = 0.0f;
}

__global__ void k_f32_to_bf16(const float* __restrict__ in, __bf16* __restrict__ out, int n) {
    int i = blockIdx.x * 256 + threadIdx.x;
    if (i < n) out[i] = (__bf16)in[i];
}

__global__ void k_embed(const int* __restrict__ idx, const float* __restrict__ emb,
                        float* __restrict__ x) {
    int row = blockIdx.x;
    size_t src = (size_t)idx[row] * kC;
    size_t dst = (size_t)row * kC;
    for (int c = threadIdx.x; c < kC; c += 256) x[dst + c] = emb[src + c];
}

// LayerNorm over C=1024; one block (256 threads) per row. Writes f32 + bf16.
__global__ __launch_bounds__(256) void k_ln(const float* __restrict__ xin,
                                            const float* __restrict__ w,
                                            const float* __restrict__ b,
                                            float* __restrict__ xf,
                                            __bf16* __restrict__ xh) {
    int row = blockIdx.x;
    int tid = threadIdx.x;
    const float* xr = xin + (size_t)row * kC;
    float vals[kC / 256];
    float s = 0.f, s2 = 0.f;
#pragma unroll
    for (int i = 0; i < kC / 256; ++i) {
        float v = xr[tid + i * 256];
        vals[i] = v;
        s += v;
        s2 += v * v;
    }
    __shared__ float rs[256], rq[256];
    rs[tid] = s; rq[tid] = s2;
    __syncthreads();
    for (int off = 128; off > 0; off >>= 1) {
        if (tid < off) { rs[tid] += rs[tid + off]; rq[tid] += rq[tid + off]; }
        __syncthreads();
    }
    float mean = rs[0] * (1.0f / kC);
    float var  = rq[0] * (1.0f / kC) - mean * mean;
    float inv  = rsqrtf(var + 1e-5f);
#pragma unroll
    for (int i = 0; i < kC / 256; ++i) {
        int c = tid + i * 256;
        float o = (vals[i] - mean) * inv * w[c] + b[c];
        xf[(size_t)row * kC + c] = o;
        xh[(size_t)row * kC + c] = (__bf16)o;
    }
}

// token-shift mix: kk = xx + (xx - xsh) * x_k  (bf16 out, feeds ffn_K GEMM)
__global__ void k_kk(const float* __restrict__ xx, const float* __restrict__ prev,
                     const float* __restrict__ xk, __bf16* __restrict__ kk) {
    int row = blockIdx.x;
    int b = row / kT, t = row % kT;
    const float* cur = xx + (size_t)row * kC;
    const float* sh  = (t == 0) ? (prev + (size_t)b * kC) : (cur - kC);
    for (int c = threadIdx.x; c < kC; c += 256) {
        float xc = cur[c];
        kk[(size_t)row * kC + c] = (__bf16)(xc + (xc - sh[c]) * xk[c]);
    }
}

__global__ void k_prev(const float* __restrict__ xx, float* __restrict__ prev) {
    int i = blockIdx.x * 256 + threadIdx.x;
    if (i < kB * kC) {
        int b = i / kC, c = i % kC;
        prev[i] = xx[((size_t)b * kT + (kT - 1)) * kC + c];
    }
}

// ======================================================================
// Chunked bidirectional WKV via WMMA (chunk Tc=32).
// grid = 2*B*H blocks, 128 threads (4 waves). Wave j owns the 64x16
// state column S[:, j*16..] as fp32 WMMA accumulators (4 tiles).
// Per chunk:  P = Rw*Ki^T (masked sigma<tau), Y = Rw*S0 + Pm*V (+u term),
//             S1 = w^32 (.) S0 + Kw^T*V   (C-operand keeps state in f32)
// ======================================================================
__global__ __launch_bounds__(128) void k_wkv_wmma(const float* __restrict__ r,
                                                  const float* __restrict__ k,
                                                  const float* __restrict__ v,
                                                  const float* __restrict__ w,
                                                  const float* __restrict__ u,
                                                  float* __restrict__ y) {
    const int dir = blockIdx.x / (kB * kH);
    const int bh  = blockIdx.x % (kB * kH);
    const int b = bh / kH, h = bh % kH;
    const int tid  = threadIdx.x;
    const int lane = tid & 31;
    const int wave = tid >> 5;     // 0..3 : m-column tile
    const int half = lane >> 4;
    const int ml   = lane & 15;

    __shared__ __align__(16) float  pw[33][64];   // w^tau
    __shared__ __align__(16) float  ip[33][64];   // w^-tau
    __shared__ __align__(16) float  vraw[32][64];
    __shared__ __align__(16) __bf16 Rw[32][64];   // A: [tau][n]
    __shared__ __align__(16) __bf16 Ki[32][64];   // B: [sigma][n]
    __shared__ __align__(16) __bf16 KwT[64][32];  // A: [n][tau]
    __shared__ __align__(16) __bf16 VT[64][32];   // B: [m][tau]
    __shared__ __align__(16) __bf16 ST[64][64];   // B: [m][n]  (S0 in bf16)
    __shared__ __align__(16) __bf16 Pm[32][32];   // A: [tau][sigma] masked
    __shared__ float uv[64];
    __shared__ float selfp[32][4];
    __shared__ float selfd[32];

    if (tid < 64) {
        float wn = w[h * kN + tid];
        uv[tid] = u[h * kN + tid];
        float pwv = 1.f, ipv = 1.f, iw = 1.f / wn;
        for (int t2 = 0; t2 <= 32; ++t2) {
            pw[t2][tid] = pwv;
            ip[t2][tid] = ipv;
            pwv *= wn;
            ipv *= iw;
        }
    }
    v8f Sacc[4];
#pragma unroll
    for (int ni = 0; ni < 4; ++ni)
        Sacc[ni] = (v8f){0.f, 0.f, 0.f, 0.f, 0.f, 0.f, 0.f, 0.f};
    __syncthreads();

    const int tau = tid >> 2;        // 0..31 (staging role)
    const int nb  = (tid & 3) * 16;  // n chunk base

    for (int s0 = 0; s0 < kT; s0 += 32) {
        // ---------------- phase A: stage chunk + export S0 (bf16) --------
        {
            int t = dir ? (kT - 1 - (s0 + tau)) : (s0 + tau);
            size_t gb = ((size_t)b * kT + t) * kC + (size_t)h * kN + nb;
            float sp = 0.f;
#pragma unroll
            for (int q = 0; q < 4; ++q) {
                union { float4 f4; float f[4]; } rr, kk2, vv;
                rr.f4  = *reinterpret_cast<const float4*>(r + gb + q * 4);
                kk2.f4 = *reinterpret_cast<const float4*>(k + gb + q * 4);
                vv.f4  = *reinterpret_cast<const float4*>(v + gb + q * 4);
#pragma unroll
                for (int e = 0; e < 4; ++e) {
                    int n = nb + q * 4 + e;
                    float rf = rr.f[e], kf = kk2.f[e], vf = vv.f[e];
                    Rw[tau][n]  = (__bf16)(rf * pw[tau][n]);
                    Ki[tau][n]  = (__bf16)(kf * ip[tau + 1][n]);
                    KwT[n][tau] = (__bf16)(kf * pw[31 - tau][n]);
                    VT[n][tau]  = (__bf16)vf;
                    vraw[tau][n] = vf;
                    sp += rf * uv[n] * kf;
                }
            }
            selfp[tau][tid & 3] = sp;
        }
        // export state (chunk-start S0) to bf16 B-layout ST[m][n]
#pragma unroll
        for (int ni = 0; ni < 4; ++ni)
#pragma unroll
            for (int jj = 0; jj < 8; ++jj)
                ST[wave * 16 + ml][ni * 16 + jj + 8 * half] = (__bf16)Sacc[ni][jj];
        __syncthreads();

        // ---------------- phase B: P tiles + inter-chunk Y ---------------
        if (tid < 32)
            selfd[tid] = selfp[tid][0] + selfp[tid][1] + selfp[tid][2] + selfp[tid][3];

        const int ti = wave >> 1, tj = wave & 1;
        v8f pacc = (v8f){0.f, 0.f, 0.f, 0.f, 0.f, 0.f, 0.f, 0.f};
#pragma unroll
        for (int ks = 0; ks < 2; ++ks) {
            FragBf a, bb;
            const __bf16* ra = &Rw[ti * 16 + ml][ks * 32];
            a.q[0] = *reinterpret_cast<const uint4*>(ra + half * 16);
            a.q[1] = *reinterpret_cast<const uint4*>(ra + half * 16 + 8);
            const __bf16* rb = &Ki[tj * 16 + ml][ks * 32 + half * 16];
            bb.q[0] = *reinterpret_cast<const uint4*>(rb);
            bb.q[1] = *reinterpret_cast<const uint4*>(rb + 8);
            pacc = __builtin_amdgcn_wmma_f32_16x16x32_bf16(false, a.v, false, bb.v,
                                                           (short)0, pacc, false, false);
        }
        v8f yacc[2];
#pragma unroll
        for (int t2 = 0; t2 < 2; ++t2)
            yacc[t2] = (v8f){0.f, 0.f, 0.f, 0.f, 0.f, 0.f, 0.f, 0.f};
#pragma unroll
        for (int ti2 = 0; ti2 < 2; ++ti2)
#pragma unroll
            for (int ks = 0; ks < 2; ++ks) {
                FragBf a, bb;
                const __bf16* ra = &Rw[ti2 * 16 + ml][ks * 32];
                a.q[0] = *reinterpret_cast<const uint4*>(ra + half * 16);
                a.q[1] = *reinterpret_cast<const uint4*>(ra + half * 16 + 8);
                const __bf16* rb = &ST[wave * 16 + ml][ks * 32 + half * 16];
                bb.q[0] = *reinterpret_cast<const uint4*>(rb);
                bb.q[1] = *reinterpret_cast<const uint4*>(rb + 8);
                yacc[ti2] = __builtin_amdgcn_wmma_f32_16x16x32_bf16(
                    false, a.v, false, bb.v, (short)0, yacc[ti2], false, false);
            }
        // masked P -> LDS (strict lower: sigma < tau)
#pragma unroll
        for (int jj = 0; jj < 8; ++jj) {
            int pr = ti * 16 + jj + 8 * half, pc = tj * 16 + ml;
            Pm[pr][pc] = (__bf16)((pc < pr) ? pacc[jj] : 0.f);
        }
        __syncthreads();

        // ---------------- phase C: intra Y, state update, output ---------
        FragBf bv;  // VT column for this wave: B frag over K=tau
        {
            const __bf16* rb = &VT[wave * 16 + ml][half * 16];
            bv.q[0] = *reinterpret_cast<const uint4*>(rb);
            bv.q[1] = *reinterpret_cast<const uint4*>(rb + 8);
        }
#pragma unroll
        for (int ti2 = 0; ti2 < 2; ++ti2) {
            FragBf a;
            const __bf16* ra = &Pm[ti2 * 16 + ml][0];
            a.q[0] = *reinterpret_cast<const uint4*>(ra + half * 16);
            a.q[1] = *reinterpret_cast<const uint4*>(ra + half * 16 + 8);
            yacc[ti2] = __builtin_amdgcn_wmma_f32_16x16x32_bf16(
                false, a.v, false, bv.v, (short)0, yacc[ti2], false, false);
        }
#pragma unroll
        for (int ni = 0; ni < 4; ++ni) {
#pragma unroll
            for (int jj = 0; jj < 8; ++jj)
                Sacc[ni][jj] *= pw[32][ni * 16 + jj + 8 * half];  // w^Tc (.) S0
            FragBf ak;
            const __bf16* ra = &KwT[ni * 16 + ml][0];
            ak.q[0] = *reinterpret_cast<const uint4*>(ra + half * 16);
            ak.q[1] = *reinterpret_cast<const uint4*>(ra + half * 16 + 8);
            Sacc[ni] = __builtin_amdgcn_wmma_f32_16x16x32_bf16(
                false, ak.v, false, bv.v, (short)0, Sacc[ni], false, false);
        }
        // output: y += Yinter + Yintra (+ self term on forward pass)
#pragma unroll
        for (int ti2 = 0; ti2 < 2; ++ti2)
#pragma unroll
            for (int jj = 0; jj < 8; ++jj) {
                int tg = ti2 * 16 + jj + 8 * half;
                int m  = wave * 16 + ml;
                int s  = s0 + tg;
                int t  = dir ? (kT - 1 - s) : s;
                float yv = yacc[ti2][jj];
                if (dir == 0) yv += selfd[tg] * vraw[tg][m];
                atomicAdd(&y[((size_t)b * kT + t) * kC + (size_t)h * kN + m], yv);
            }
        __syncthreads();
    }
}

// ======================================================================
// WMMA GEMM: C[m][n] = epilogue( sum_k A[m][k]*B[k][n] )
//   A tile (bf16, 128x32, padded LDS rows of 40 bf16) fetched by the
//   Tensor Data Mover (tensor_load_to_lds, pad_interval=16dw/pad=4dw).
//   B: fp32 weights converted+transposed into LDS Bt[n][k] by the block.
// MODE 0: outF = v; 1: outF += v; 2: outH = relu(v)^2 (bf16);
// MODE 3: outF += v * enn[idx[row]*N + col]
// ======================================================================
constexpr int BM = 128, BN = 128, BK = 32;
constexpr int SA = 40;  // padded LDS row stride (bf16 elems), keeps 16B align

template <int MODE>
__global__ __launch_bounds__(256) void k_gemm(const __bf16* __restrict__ A,
                                              const float* __restrict__ Bw,
                                              float* __restrict__ outF,
                                              __bf16* __restrict__ outH,
                                              const float* __restrict__ enn,
                                              const int* __restrict__ idx,
                                              int M, int Nn, int K) {
    __shared__ __align__(16) __bf16 As[BM * SA];
    __shared__ __align__(16) __bf16 Bs[BN * SA];

    const int tid  = threadIdx.x;
    const int lane = tid & 31;
    const int wave = tid >> 5;
    const int wr = wave >> 1;  // wave row 0..3  (32 rows each)
    const int wc = wave & 1;   // wave col 0..1  (64 cols each)
    const int m0 = blockIdx.y * BM;
    const int n0 = blockIdx.x * BN;
    const unsigned ldsA = (unsigned)(size_t)(&As[0]);  // LDS byte offset

    v8f acc[2][4];
#pragma unroll
    for (int i = 0; i < 2; ++i)
#pragma unroll
        for (int j = 0; j < 4; ++j) acc[i][j] = (v8f){0.f, 0.f, 0.f, 0.f, 0.f, 0.f, 0.f, 0.f};

    const int bK  = tid >> 5;        // 0..7  (four passes: +0,8,16,24)
    const int bN4 = (tid & 31) * 4;  // 4-float chunk within 128 n

    const int half = lane >> 4;  // lane group 0/1
    const int ml   = lane & 15;

    for (int k0 = 0; k0 < K; k0 += BK) {
        // ---- A tile via Tensor Data Mover (wave 0 issues the DMA) ----
        if (wave == 0)
            tdm_load_2d_bf16(ldsA, A + (size_t)m0 * K + k0, BK, BM,
                             (unsigned long long)K,
                             /*pad_interval 16 dwords*/ 3, /*pad 4 dwords*/ 3);
        if (k0 + BK < K)  // hint next B tile toward L2 (global_prefetch_b8)
            __builtin_prefetch(Bw + (size_t)(k0 + BK + bK) * Nn + n0 + bN4, 0, 1);
        // ---- stage B tile transposed+converted: Bs[n][k] (bf16) ----
#pragma unroll
        for (int q = 0; q < 4; ++q) {
            int kk2 = bK + q * 8;
            const float4 f =
                *reinterpret_cast<const float4*>(Bw + (size_t)(k0 + kk2) * Nn + n0 + bN4);
            Bs[(bN4 + 0) * SA + kk2] = (__bf16)f.x;
            Bs[(bN4 + 1) * SA + kk2] = (__bf16)f.y;
            Bs[(bN4 + 2) * SA + kk2] = (__bf16)f.z;
            Bs[(bN4 + 3) * SA + kk2] = (__bf16)f.w;
        }
        if (wave == 0) __builtin_amdgcn_s_wait_tensorcnt(0);
        __syncthreads();

        // ---- build fragments (2x ds_load_b128 each) ----
        FragBf af[2], bfr[4];
#pragma unroll
        for (int fm = 0; fm < 2; ++fm) {
            const __bf16* rp = &As[(wr * 32 + fm * 16 + ml) * SA];
            af[fm].q[0] = *reinterpret_cast<const uint4*>(rp + half * 16);
            af[fm].q[1] = *reinterpret_cast<const uint4*>(rp + half * 16 + 8);
        }
#pragma unroll
        for (int fn = 0; fn < 4; ++fn) {
            const __bf16* rp = &Bs[(wc * 64 + fn * 16 + ml) * SA + half * 16];
            bfr[fn].q[0] = *reinterpret_cast<const uint4*>(rp);
            bfr[fn].q[1] = *reinterpret_cast<const uint4*>(rp + 8);
        }
        // ---- 8x v_wmma_f32_16x16x32_bf16 per K-step ----
#pragma unroll
        for (int fm = 0; fm < 2; ++fm)
#pragma unroll
            for (int fn = 0; fn < 4; ++fn)
                acc[fm][fn] = __builtin_amdgcn_wmma_f32_16x16x32_bf16(
                    false, af[fm].v, false, bfr[fn].v, (short)0, acc[fm][fn], false, false);
        __syncthreads();
    }

    // ---- epilogue: C/D layout = VGPR j -> row (j + 8*half), col = lane&15 ----
#pragma unroll
    for (int fm = 0; fm < 2; ++fm) {
#pragma unroll
        for (int fn = 0; fn < 4; ++fn) {
            int rbase = m0 + wr * 32 + fm * 16 + half * 8;
            int col   = n0 + wc * 64 + fn * 16 + ml;
#pragma unroll
            for (int j = 0; j < 8; ++j) {
                int row = rbase + j;
                float vv = acc[fm][fn][j];
                size_t o = (size_t)row * Nn + col;
                if (MODE == 0) {
                    outF[o] = vv;
                } else if (MODE == 1) {
                    outF[o] += vv;
                } else if (MODE == 2) {
                    float t = vv > 0.f ? vv : 0.f;
                    outH[o] = (__bf16)(t * t);
                } else {  // MODE == 3
                    outF[o] += vv * enn[(size_t)idx[row] * Nn + col];
                }
            }
        }
    }
}

// ======================================================================
// launch
// ======================================================================
extern "C" void kernel_launch(void* const* d_in, const int* in_sizes, int n_in,
                              void* d_out, int out_size, void* d_ws, size_t ws_size,
                              hipStream_t stream) {
    (void)in_sizes; (void)n_in; (void)out_size; (void)ws_size;

    const int*   idx     = (const int*)  d_in[0];
    const float* emb     = (const float*)d_in[1];
    const float* ln1_w   = (const float*)d_in[2];
    const float* ln1_b   = (const float*)d_in[3];
    const float* w_r     = (const float*)d_in[4];
    const float* w_k     = (const float*)d_in[5];
    const float* w_v     = (const float*)d_in[6];
    const float* decay   = (const float*)d_in[7];
    const float* uu      = (const float*)d_in[8];
    const float* w_o     = (const float*)d_in[9];
    const float* ln2_w   = (const float*)d_in[10];
    const float* ln2_b   = (const float*)d_in[11];
    const float* x_k     = (const float*)d_in[12];
    const float* ffn_K   = (const float*)d_in[13];
    const float* ffn_V   = (const float*)d_in[14];
    const float* ffn_enn = (const float*)d_in[15];
    const float* lno_w   = (const float*)d_in[16];
    const float* lno_b   = (const float*)d_in[17];
    const float* head_w  = (const float*)d_in[18];
    float* out = (float*)d_out;

    char* wp = (char*)d_ws;
    auto alloc = [&](size_t bytes) -> void* {
        void* p = (void*)wp;
        wp += (bytes + 255) & ~(size_t)255;
        return p;
    };
    float*  x    = (float*) alloc((size_t)kM * kC * 4);
    float*  xx   = (float*) alloc((size_t)kM * kC * 4);
    __bf16* xxh  = (__bf16*)alloc((size_t)kM * kC * 2);
    float*  rb   = (float*) alloc((size_t)kM * kC * 4);
    float*  kb   = (float*) alloc((size_t)kM * kC * 4);
    float*  vb   = (float*) alloc((size_t)kM * kC * 4);
    float*  yb   = (float*) alloc((size_t)kM * kC * 4);
    __bf16* ybh  = (__bf16*)alloc((size_t)kM * kC * 2);
    __bf16* kkh  = (__bf16*)alloc((size_t)kM * kC * 2);
    __bf16* hh   = (__bf16*)alloc((size_t)kM * kFF * 2);
    float*  prev = (float*) alloc((size_t)kB * kC * 4);

    const dim3 gC(kC / BN, kM / BM);
    const dim3 gF(kFF / BN, kM / BM);
    const dim3 gH(kV / BN, kM / BM);

    k_embed<<<kM, 256, 0, stream>>>(idx, emb, x);
    k_zero_f32<<<(kB * kC + 255) / 256, 256, 0, stream>>>(prev, kB * kC);

    for (int i = 0; i < kL; ++i) {
        const size_t oc = (size_t)i * kC;
        k_ln<<<kM, 256, 0, stream>>>(x, ln1_w + oc, ln1_b + oc, xx, xxh);
        k_gemm<0><<<gC, 256, 0, stream>>>(xxh, w_r + (size_t)i * kC * kC, rb, nullptr,
                                          nullptr, nullptr, kM, kC, kC);
        k_gemm<0><<<gC, 256, 0, stream>>>(xxh, w_k + (size_t)i * kC * kC, kb, nullptr,
                                          nullptr, nullptr, kM, kC, kC);
        k_gemm<0><<<gC, 256, 0, stream>>>(xxh, w_v + (size_t)i * kC * kC, vb, nullptr,
                                          nullptr, nullptr, kM, kC, kC);
        k_zero_f32<<<(kM * kC + 255) / 256, 256, 0, stream>>>(yb, kM * kC);
        k_wkv_wmma<<<2 * kB * kH, 128, 0, stream>>>(rb, kb, vb, decay + (size_t)i * kH * kN,
                                                    uu + (size_t)i * kH * kN, yb);
        k_f32_to_bf16<<<(kM * kC + 255) / 256, 256, 0, stream>>>(yb, ybh, kM * kC);
        k_gemm<1><<<gC, 256, 0, stream>>>(ybh, w_o + (size_t)i * kC * kC, x, nullptr,
                                          nullptr, nullptr, kM, kC, kC);
        k_ln<<<kM, 256, 0, stream>>>(x, ln2_w + oc, ln2_b + oc, xx, xxh);
        k_kk<<<kM, 256, 0, stream>>>(xx, prev, x_k + oc, kkh);
        k_prev<<<(kB * kC + 255) / 256, 256, 0, stream>>>(xx, prev);
        k_gemm<2><<<gF, 256, 0, stream>>>(kkh, ffn_K + (size_t)i * kC * kFF, nullptr, hh,
                                          nullptr, nullptr, kM, kFF, kC);
        k_gemm<3><<<gC, 256, 0, stream>>>(hh, ffn_V + (size_t)i * kFF * kC, x, nullptr,
                                          ffn_enn + (size_t)i * kV * kC, idx, kM, kC, kFF);
    }
    k_ln<<<kM, 256, 0, stream>>>(x, lno_w, lno_b, xx, xxh);
    k_gemm<0><<<gH, 256, 0, stream>>>(xxh, head_w, out, nullptr, nullptr, nullptr,
                                      kM, kV, kC);
}
